// EinsumGDN_68942815035712
// MI455X (gfx1250) — matrix operand order repaired and checked by reading, not compiled, not run on hardware
//
#include <hip/hip_runtime.h>

// ---------------------------------------------------------------------------
// GDN (compressai semantics) for x[4,192,256,256] fp32.
//   denom[i,p] = sum_c gamma_eff[i,c] * x2[c,p] ;  out = x * rsqrt(denom+beta_eff)
// GEMM M=192,K=192,N=262144 -> split-bf16 WMMA (v_wmma_f32_16x16x32_bf16),
// 3-product hi/lo emulation for ~fp32 accuracy at bf16 matrix throughput.
// B-tile global loads are issued as an independent clause BEFORE the LDS
// staging/barriers so HBM latency overlaps gamma staging.
// ---------------------------------------------------------------------------

typedef __attribute__((ext_vector_type(16))) __bf16 v16bf;
typedef __attribute__((ext_vector_type(8)))  float  v8f;

#define C_DIM   192
#define HW_DIM  65536          // 256*256
#define N_IMG   4
#define MT      12             // 192/16 output-channel tiles
#define KT      6              // 192/32 K tiles
#define PIX_TOTAL (N_IMG * HW_DIM)          // 262144 pixels
#define WAVES_PER_BLOCK 8
#define PIX_PER_BLOCK  (WAVES_PER_BLOCK * 16)   // 128
#define NUM_BLOCKS (PIX_TOTAL / PIX_PER_BLOCK)  // 2048

// NonNegativeParametrizer constants
#define PEDESTAL      1.4551915228366852e-11f   // 2^-36
#define BOUND_BETA    1.0000072759191477e-3f    // sqrt(1e-6 + 2^-36)
#define BOUND_GAMMA   3.814697265625e-6f        // 2^-18

__global__ __launch_bounds__(256) void gdn_wmma_kernel(
    const float* __restrict__ x,
    const float* __restrict__ beta,
    const float* __restrict__ gamma,
    float* __restrict__ out)
{
    // gamma_eff hi/lo for ONE K-slice (192 x 32), stored in WMMA 16-bit
    // A-matrix lane layout: element index = mt*512 + lane*16 + j
    __shared__ __align__(64) __bf16 sGhi[MT * 512];
    __shared__ __align__(64) __bf16 sGlo[MT * 512];

    const int tid     = threadIdx.x;
    const int wave    = tid >> 5;
    const int lane    = tid & 31;
    const int lane15  = lane & 15;
    const int hihalf  = lane >> 4;        // 0: lanes 0-15, 1: lanes 16-31

    const int tile  = blockIdx.x * WAVES_PER_BLOCK + wave; // 16-pixel tile id
    const int pix   = tile * 16;
    const int n_img = pix >> 16;          // / HW_DIM
    const int p0    = pix & (HW_DIM - 1);

    const float* __restrict__ xn = x   + (size_t)n_img * C_DIM * HW_DIM;
    float*       __restrict__ on = out + (size_t)n_img * C_DIM * HW_DIM;

    v8f zero = {};
    v8f acc[MT];
    #pragma unroll
    for (int m = 0; m < MT; ++m) acc[m] = zero;

    for (int kt = 0; kt < KT; ++kt) {
        // ---- B tile global loads FIRST: independent clause, overlaps the
        // gamma staging + barriers below. 16-bit B 32x16 layout: lane = N
        // column (l&15); halves j hold K = j<8 ? 8h+j : 16+8h+(j-8), h=l>>4
        float xv[16];
        const float* __restrict__ xb =
            xn + (size_t)(kt * 32 + hihalf * 8) * HW_DIM + p0 + lane15;
        #pragma unroll
        for (int j = 0; j < 16; ++j) {
            xv[j] = xb[(size_t)(j + (j >= 8 ? 8 : 0)) * HW_DIM];   // coalesced across half-wave
        }

        // ---- stage gamma_eff K-slice into LDS (bf16 hi/lo, A layout) ----
        __syncthreads();   // previous iteration's ds_loads done before rewrite
        #pragma unroll
        for (int it = 0; it < (MT * 512) / 256; ++it) {   // 24 iters
            int t  = tid + it * 256;           // 0 .. 6143
            int mt = t >> 9;
            int r  = t & 511;
            int l  = r >> 4;                   // target lane
            int j  = r & 15;                   // target half within v16bf
            // ISA 16-bit A 16x32 layout: lanes<16: K = j<8 ? j : j+8
            //                            lanes>=16: K = j<8 ? j+8 : j+16
            int K  = (j < 8) ? (j + ((l < 16) ? 0 : 8))
                             : (j + ((l < 16) ? 8 : 16));
            int i  = mt * 16 + (l & 15);       // output channel (M)
            int c  = kt * 32 + K;              // reduced channel (K)
            float g  = gamma[i * C_DIM + c];
            g        = fmaxf(g, BOUND_GAMMA);
            float ge = g * g - PEDESTAL;       // gamma_eff
            __bf16 h  = (__bf16)ge;
            __bf16 lo = (__bf16)(ge - (float)h);
            sGhi[t] = h;
            sGlo[t] = lo;
        }
        __syncthreads();

        // ---- convert B tile: x2 split into bf16 hi/lo ----
        v16bf bhi, blo;
        #pragma unroll
        for (int j = 0; j < 16; ++j) {
            float x2 = xv[j] * xv[j];
            __bf16 h = (__bf16)x2;
            bhi[j] = h;
            blo[j] = (__bf16)(x2 - (float)h);
        }

        // ---- 12 output tiles, 3-product split-bf16 accumulate ----
        #pragma unroll
        for (int mt = 0; mt < MT; ++mt) {
            v16bf ahi = *(const v16bf*)&sGhi[mt * 512 + lane * 16];
            v16bf alo = *(const v16bf*)&sGlo[mt * 512 + lane * 16];
            acc[mt] = __builtin_amdgcn_wmma_f32_16x16x32_bf16(
                false, ahi, false, bhi, (short)0, acc[mt], false, false);
            acc[mt] = __builtin_amdgcn_wmma_f32_16x16x32_bf16(
                false, ahi, false, blo, (short)0, acc[mt], false, false);
            acc[mt] = __builtin_amdgcn_wmma_f32_16x16x32_bf16(
                false, alo, false, bhi, (short)0, acc[mt], false, false);
        }
    }

    // ---- epilogue: + beta_eff, rsqrt, scale by x, store ----
    // C/D 16x16 f32 layout: lane 0-15 -> N=lane (pixel), VGPR r -> M=r (+8 for
    // lanes 16-31).
    #pragma unroll
    for (int mt = 0; mt < MT; ++mt) {
        #pragma unroll
        for (int r = 0; r < 8; ++r) {
            int i = mt * 16 + r + (hihalf ? 8 : 0);
            float bv = fmaxf(beta[i], BOUND_BETA);
            float be = bv * bv - PEDESTAL;               // beta_eff
            float d  = acc[mt][r] + be;
            float inv = __builtin_amdgcn_rsqf(d);        // v_rsq_f32
            size_t off = (size_t)i * HW_DIM + p0 + lane15;
            on[off] = xn[off] * inv;
        }
    }
}

extern "C" void kernel_launch(void* const* d_in, const int* in_sizes, int n_in,
                              void* d_out, int out_size, void* d_ws, size_t ws_size,
                              hipStream_t stream) {
    (void)in_sizes; (void)n_in; (void)out_size; (void)d_ws; (void)ws_size;
    const float* x     = (const float*)d_in[0];
    const float* beta  = (const float*)d_in[1];
    const float* gamma = (const float*)d_in[2];
    float* out = (float*)d_out;

    gdn_wmma_kernel<<<NUM_BLOCKS, 256, 0, stream>>>(x, beta, gamma, out);
}
